// Fringed_node_align_Node_loss_34505767256120
// MI455X (gfx1250) — matrix operand order repaired and checked by reading, not compiled, not run on hardware
//
#include <hip/hip_runtime.h>
#include <hip/hip_bf16.h>
#include <stdint.h>

// ---------------------------------------------------------------------------
// Types / WMMA helpers (CDNA5 bf16 WMMA, wave32)
// ---------------------------------------------------------------------------
typedef unsigned short u16;
typedef __attribute__((ext_vector_type(16))) __bf16 v16bf;
typedef __attribute__((ext_vector_type(8)))  float  v8f;
typedef __attribute__((ext_vector_type(8)))  u16    v8u;

struct Frag16 { v8u lo, hi; };   // 16 bf16 per lane = one WMMA A/B operand

static __device__ __forceinline__ u16 f2bfu(float f) {
  unsigned u = __float_as_uint(f);
  return (u16)((u + 0x7FFFu + ((u >> 16) & 1u)) >> 16);   // RNE f32->bf16
}

static __device__ __forceinline__ v8u cvt8(const float* s) {
  v8u r;
#pragma unroll
  for (int i = 0; i < 8; ++i) r[i] = f2bfu(s[i]);
  return r;
}

static __device__ __forceinline__ v8f wmma_bf16(Frag16 a, Frag16 b, v8f c) {
  return __builtin_amdgcn_wmma_f32_16x16x32_bf16(
      false, __builtin_bit_cast(v16bf, a),
      false, __builtin_bit_cast(v16bf, b),
      (short)0, c, false, false);
}

// Pre-swizzled weight tile: 32 lanes x 16 bf16 contiguous (1024 B / tile).
static __device__ __forceinline__ Frag16 loadW(const u16* Ws, int tile, int lane) {
  const v8u* p = reinterpret_cast<const v8u*>(Ws + (size_t)tile * 512 + lane * 16);
  Frag16 f; f.lo = p[0]; f.hi = p[1];
  return f;
}

#define ZERO8 {0.f,0.f,0.f,0.f,0.f,0.f,0.f,0.f}

// ---------------------------------------------------------------------------
// Problem constants (match reference)
// ---------------------------------------------------------------------------
#define NUM_GRAPHS 256
#define MAX_SET    128
#define DD         64
#define OUTER_ITERS 5
#define SINK_ITERS 10
#define TEMP_INV   10.0f   // 1/0.1

// ---------------------------------------------------------------------------
// Meta: SIZES/OFFSETS/NODE_SLOT/NODE_POS from closed form; mask=1
// ---------------------------------------------------------------------------
__global__ void meta_kernel(int* offsets, int* node_slot, int* node_pos,
                            float* mask, int N) {
  __shared__ int sz[256], off[256];
  int g = threadIdx.x;
  int s = 96 + (13 * g) % 33;
  sz[g] = s;
  __syncthreads();
  if (g == 0) { int a = 0; for (int i = 0; i < 256; ++i) { off[i] = a; a += sz[i]; } }
  __syncthreads();
  offsets[g] = off[g];
  int base = off[g];
  for (int i = 0; i < s; ++i) {
    node_slot[base + i] = g;
    node_pos [base + i] = i;
    mask     [base + i] = 1.0f;
  }
  (void)N;
}

// ---------------------------------------------------------------------------
// Encoders: h_enc = nf @ W_ne + b (f32 + bf16), e_bf = bf16(ef @ W_ee + b)
// ---------------------------------------------------------------------------
__global__ void enc_nodes_kernel(const float* nf, const float* W, const float* b,
                                 float* h_enc, u16* hb, long total) {
  long i = (long)blockIdx.x * blockDim.x + threadIdx.x;
  if (i >= total) return;
  long n = i >> 6; int d = (int)(i & 63);
  float a = b[d];
  for (int k = 0; k < 32; ++k) a += nf[n * 32 + k] * W[k * 64 + d];
  h_enc[i] = a; hb[i] = f2bfu(a);
}

__global__ void enc_edges_kernel(const float* ef, const float* W, const float* b,
                                 u16* ebf, long total) {
  long i = (long)blockIdx.x * blockDim.x + threadIdx.x;
  if (i >= total) return;
  long e = i >> 6; int d = (int)(i & 63);
  float a = b[d];
  for (int k = 0; k < 16; ++k) a += ef[e * 16 + k] * W[k * 64 + d];
  ebf[i] = f2bfu(a);
}

// ---------------------------------------------------------------------------
// Build fused GRU weight  Wcat (192 x 256) and b_cat(256):
// cols 0..63 : r-sum, 64..127 : z-sum, 128..191 : i_n, 192..255 : h_n
// ---------------------------------------------------------------------------
__global__ void build_wcat_kernel(const float* W_ih, const float* W_hh,
                                  const float* b_ih, const float* b_hh,
                                  float* Wcat, float* b_cat) {
  int i = blockIdx.x * blockDim.x + threadIdx.x;
  if (i >= 192 * 256) return;
  int k = i >> 8, c = i & 255;
  int grp = c >> 6, d = c & 63;
  float v = 0.f;
  if (grp == 0) v = (k < 128) ? W_ih[k * 192 + d]        : W_hh[(k - 128) * 192 + d];
  if (grp == 1) v = (k < 128) ? W_ih[k * 192 + 64 + d]   : W_hh[(k - 128) * 192 + 64 + d];
  if (grp == 2) v = (k < 128) ? W_ih[k * 192 + 128 + d]  : 0.f;
  if (grp == 3) v = (k < 128) ? 0.f                      : W_hh[(k - 128) * 192 + 128 + d];
  Wcat[i] = v;
  if (k == 0) {
    float bb = 0.f;
    if (grp == 0) bb = b_ih[d]        + b_hh[d];
    if (grp == 1) bb = b_ih[64 + d]   + b_hh[64 + d];
    if (grp == 2) bb = b_ih[128 + d];
    if (grp == 3) bb = b_hh[128 + d];
    b_cat[c] = bb;
  }
}

// ---------------------------------------------------------------------------
// Swizzle f32 weight (K x Ncols, row-major) into bf16 B-fragment tiles.
// elem i of lane l in tile (kt,nt):  K = kt*32 + (l<16?0:8) + (i<8?i:i+8) , N = nt*16 + (l&15)
// ---------------------------------------------------------------------------
__global__ void swizzle_kernel(const float* W, u16* out, int K, int Ncols) {
  int total = (K >> 5) * (Ncols >> 4) * 512;
  int gid = blockIdx.x * blockDim.x + threadIdx.x;
  if (gid >= total) return;
  int tile = gid >> 9, r = gid & 511, l = r >> 4, i = r & 15;
  int ntiles = Ncols >> 4;
  int kt = tile / ntiles, nt = tile % ntiles;
  int k = kt * 32 + ((l < 16) ? 0 : 8) + ((i < 8) ? i : i + 8);
  int n = nt * 16 + (l & 15);
  out[gid] = f2bfu(W[k * Ncols + n]);
}

// ---------------------------------------------------------------------------
// Zero helpers
// ---------------------------------------------------------------------------
__global__ void zero_f32_kernel(float* p, long n) {
  long i = (long)blockIdx.x * blockDim.x + threadIdx.x;
  if (i < n) p[i] = 0.f;
}
__global__ void zero_u16_kernel(u16* p, long n) {
  long i = (long)blockIdx.x * blockDim.x + threadIdx.x;
  if (i < n) p[i] = 0;
}

// ---------------------------------------------------------------------------
// Edge message MLP (one wave = 16 edges), both directions via blockIdx.y.
//   x = [h[f] | h[t] | e]  (192, bf16)
//   hid = relu(x @ Wm1 + b1)  -> staged bf16 in LDS (per-wave, stride 136)
//   m   = (hid @ Wm2 + b2) * mask[f]  -> atomicAdd into agg[t]
// __launch_bounds__(256,1): allow max VGPRs so the batched B fragments stay
// resident and loads clause ahead of the WMMA stream.
// ---------------------------------------------------------------------------
__global__ void __launch_bounds__(256, 1)
edge_mlp_kernel(const u16* __restrict__ hb, const u16* __restrict__ ebf,
                const u16* __restrict__ Wm1s, const u16* __restrict__ Wm2s,
                const float* __restrict__ b_m1, const float* __restrict__ b_m2,
                const float* __restrict__ mask,
                const int* __restrict__ from_idx, const int* __restrict__ to_idx,
                float* __restrict__ agg, int E) {
  __shared__ u16 hid[8 * 16 * 136];
  int lane = threadIdx.x & 31, wave = threadIdx.x >> 5;
  int dir = blockIdx.y;
  long tile = (long)blockIdx.x * 8 + wave;
  long e0 = tile * 16;
  if (e0 >= E) return;
  int row = lane & 15;
  long e = e0 + row;
  int valid = (e < E);
  long ec = valid ? e : (long)(E - 1);
  int f = from_idx[ec], t = to_idx[ec];
  if (dir) { int tmp = f; f = t; t = tmp; }
  float scale = valid ? mask[f] : 0.0f;
  int kb = (lane < 16) ? 0 : 8;
  u16* myHid = &hid[wave * 16 * 136];

  // ---- GEMM1: X(16x192) @ Wm1(192x128) ----
  v8f acc[8];
#pragma unroll
  for (int nt = 0; nt < 8; ++nt) { v8f z = ZERO8; acc[nt] = z; }
#pragma unroll
  for (int kt = 0; kt < 6; ++kt) {
    Frag16 a;
#pragma unroll
    for (int h = 0; h < 2; ++h) {
      int c0 = kt * 32 + kb + 16 * h;
      const u16* src;
      if (c0 < 64)        src = &hb[(long)f * 64 + c0];
      else if (c0 < 128)  src = &hb[(long)t * 64 + (c0 - 64)];
      else                src = &ebf[ec * 64 + (c0 - 128)];
      v8u v = *reinterpret_cast<const v8u*>(src);
      if (h == 0) a.lo = v; else a.hi = v;
    }
    Frag16 bfr[8];
#pragma unroll
    for (int nt = 0; nt < 8; ++nt) bfr[nt] = loadW(Wm1s, kt * 8 + nt, lane);
#pragma unroll
    for (int nt = 0; nt < 8; ++nt) acc[nt] = wmma_bf16(a, bfr[nt], acc[nt]);
  }
#pragma unroll
  for (int nt = 0; nt < 8; ++nt) {
#pragma unroll
    for (int j = 0; j < 8; ++j) {
      int rr = j + ((lane < 16) ? 0 : 8);
      int c  = nt * 16 + (lane & 15);
      float v = fmaxf(acc[nt][j] + b_m1[c], 0.f);
      myHid[rr * 136 + c] = f2bfu(v);
    }
  }
  asm volatile("" ::: "memory");   // per-wave LDS producer->consumer ordering

  // ---- GEMM2: hid(16x128) @ Wm2(128x128) ----
#pragma unroll
  for (int nt = 0; nt < 8; ++nt) { v8f z = ZERO8; acc[nt] = z; }
#pragma unroll
  for (int kt = 0; kt < 4; ++kt) {
    Frag16 a;
    int c0 = kt * 32 + kb;
    a.lo = *reinterpret_cast<const v8u*>(&myHid[row * 136 + c0]);
    a.hi = *reinterpret_cast<const v8u*>(&myHid[row * 136 + c0 + 16]);
    Frag16 bfr[8];
#pragma unroll
    for (int nt = 0; nt < 8; ++nt) bfr[nt] = loadW(Wm2s, kt * 8 + nt, lane);
#pragma unroll
    for (int nt = 0; nt < 8; ++nt) acc[nt] = wmma_bf16(a, bfr[nt], acc[nt]);
  }
  // scatter with mask * atomicAdd (segment_sum)
#pragma unroll
  for (int nt = 0; nt < 8; ++nt) {
#pragma unroll
    for (int j = 0; j < 8; ++j) {
      int rr = j + ((lane < 16) ? 0 : 8);
      if (e0 + rr < E) {
        int c = nt * 16 + (lane & 15);
        float sc = __shfl(scale, rr, 32);
        int   tt = __shfl(t, rr, 32);
        atomicAdd(&agg[(long)tt * 128 + c], (acc[nt][j] + b_m2[c]) * sc);
      }
    }
  }
}

// ---------------------------------------------------------------------------
// Fused GRU: [agg(128) | h_enc(64)] @ Wcat(192x256) + b_cat, then gates,
// result scattered directly into stacked (f32) and stackedb (bf16).
// ---------------------------------------------------------------------------
__global__ void __launch_bounds__(256, 1)
gru_kernel(const float* __restrict__ agg, const u16* __restrict__ hb,
           const float* __restrict__ h_enc,
           const u16* __restrict__ Wcats, const float* __restrict__ b_cat,
           const int* __restrict__ node_slot, const int* __restrict__ node_pos,
           float* __restrict__ stacked, u16* __restrict__ stackedb, int N) {
  int lane = threadIdx.x & 31, wave = threadIdx.x >> 5;
  long tile = (long)blockIdx.x * 8 + wave;
  long n0 = tile * 16;
  if (n0 >= N) return;
  int row = lane & 15;
  long n = n0 + row; if (n >= N) n = N - 1;
  int kb = (lane < 16) ? 0 : 8;

  v8f acc[16];
#pragma unroll
  for (int nt = 0; nt < 16; ++nt) { v8f z = ZERO8; acc[nt] = z; }
#pragma unroll
  for (int kt = 0; kt < 6; ++kt) {
    Frag16 a;
#pragma unroll
    for (int h = 0; h < 2; ++h) {
      int c0 = kt * 32 + kb + 16 * h;
      v8u v;
      if (c0 < 128) v = cvt8(&agg[n * 128 + c0]);
      else          v = *reinterpret_cast<const v8u*>(&hb[n * 64 + (c0 - 128)]);
      if (h == 0) a.lo = v; else a.hi = v;
    }
#pragma unroll
    for (int grpN = 0; grpN < 2; ++grpN) {
      Frag16 bfr[8];
#pragma unroll
      for (int q = 0; q < 8; ++q) bfr[q] = loadW(Wcats, kt * 16 + grpN * 8 + q, lane);
#pragma unroll
      for (int q = 0; q < 8; ++q) acc[grpN * 8 + q] = wmma_bf16(a, bfr[q], acc[grpN * 8 + q]);
    }
  }

  // gates + scatter: lane owns rows j+(0|8), cols (lane&15)+16*dk in every group
#pragma unroll
  for (int j = 0; j < 8; ++j) {
    long nj = n0 + j + ((lane < 16) ? 0 : 8);
    if (nj >= N) continue;
    int slot = node_slot[nj], pos = node_pos[nj];
#pragma unroll
    for (int dk = 0; dk < 4; ++dk) {
      int d = (lane & 15) + dk * 16;
      float rs  = acc[dk][j]      + b_cat[d];
      float zs  = acc[4 + dk][j]  + b_cat[64 + d];
      float inn = acc[8 + dk][j]  + b_cat[128 + d];
      float hn  = acc[12 + dk][j] + b_cat[192 + d];
      float r = 1.f / (1.f + __expf(-rs));
      float z = 1.f / (1.f + __expf(-zs));
      float nn = tanhf(inn + r * hn);
      float ho = (1.f - z) * nn + z * h_enc[nj * 64 + d];
      long o = ((long)slot * MAX_SET + pos) * DD + d;
      stacked[o]  = ho;
      stackedb[o] = f2bfu(ho);
    }
  }
}

// ---------------------------------------------------------------------------
// Per-graph MLP: t = relu(S @ Wf1 + b1) @ Wf2 + b2, masked rows -> mqc (bf16)
// One block (8 waves) per graph; wave handles its 16-row tile end-to-end.
// ---------------------------------------------------------------------------
__global__ void __launch_bounds__(256, 1)
tqtc_kernel(const u16* __restrict__ stackedb,
            const u16* __restrict__ Wf1s, const u16* __restrict__ Wf2s,
            const float* __restrict__ b_f1, const float* __restrict__ b_f2,
            u16* __restrict__ mqc) {
  __shared__ u16 T1[8 * 16 * 136];
  int lane = threadIdx.x & 31, wave = threadIdx.x >> 5;
  int g = blockIdx.x;
  int gsz = 96 + (13 * g) % 33;
  int row = lane & 15, kb = (lane < 16) ? 0 : 8;
  const u16* S = &stackedb[(long)g * MAX_SET * DD];
  u16* myT = &T1[wave * 16 * 136];
  int r0 = wave * 16;

  v8f acc[8];
#pragma unroll
  for (int nt = 0; nt < 8; ++nt) { v8f z = ZERO8; acc[nt] = z; }
#pragma unroll
  for (int kt = 0; kt < 2; ++kt) {
    Frag16 a;
    int c0 = kt * 32 + kb;
    a.lo = *reinterpret_cast<const v8u*>(&S[(r0 + row) * 64 + c0]);
    a.hi = *reinterpret_cast<const v8u*>(&S[(r0 + row) * 64 + c0 + 16]);
    Frag16 bfr[8];
#pragma unroll
    for (int nt = 0; nt < 8; ++nt) bfr[nt] = loadW(Wf1s, kt * 8 + nt, lane);
#pragma unroll
    for (int nt = 0; nt < 8; ++nt) acc[nt] = wmma_bf16(a, bfr[nt], acc[nt]);
  }
#pragma unroll
  for (int nt = 0; nt < 8; ++nt) {
#pragma unroll
    for (int j = 0; j < 8; ++j) {
      int rr = j + ((lane < 16) ? 0 : 8);
      int c  = nt * 16 + (lane & 15);
      myT[rr * 136 + c] = f2bfu(fmaxf(acc[nt][j] + b_f1[c], 0.f));
    }
  }
  asm volatile("" ::: "memory");

#pragma unroll
  for (int nt = 0; nt < 8; ++nt) { v8f z = ZERO8; acc[nt] = z; }
#pragma unroll
  for (int kt = 0; kt < 4; ++kt) {
    Frag16 a;
    int c0 = kt * 32 + kb;
    a.lo = *reinterpret_cast<const v8u*>(&myT[row * 136 + c0]);
    a.hi = *reinterpret_cast<const v8u*>(&myT[row * 136 + c0 + 16]);
    Frag16 bfr[8];
#pragma unroll
    for (int nt = 0; nt < 8; ++nt) bfr[nt] = loadW(Wf2s, kt * 8 + nt, lane);
#pragma unroll
    for (int nt = 0; nt < 8; ++nt) acc[nt] = wmma_bf16(a, bfr[nt], acc[nt]);
  }
#pragma unroll
  for (int nt = 0; nt < 8; ++nt) {
#pragma unroll
    for (int j = 0; j < 8; ++j) {
      int rr = r0 + j + ((lane < 16) ? 0 : 8);
      int c  = nt * 16 + (lane & 15);
      float v = acc[nt][j] + b_f2[c];
      if (rr >= gsz) v = 0.f;                      // row mask
      mqc[(long)g * (MAX_SET * MAX_SET) + rr * 128 + c] = f2bfu(v);
    }
  }
}

// ---------------------------------------------------------------------------
// sim[p] = mq @ mc^T  (K = feature dim 128). B = mc^T -> contiguous mc rows.
// ---------------------------------------------------------------------------
__global__ void __launch_bounds__(256, 1)
sim_kernel(const u16* __restrict__ mqc, float* __restrict__ sim) {
  int lane = threadIdx.x & 31, wave = threadIdx.x >> 5;
  int p = blockIdx.x;
  const u16* mq = &mqc[(long)(2 * p)     * (MAX_SET * MAX_SET)];
  const u16* mc = &mqc[(long)(2 * p + 1) * (MAX_SET * MAX_SET)];
  int row = lane & 15, kbA = (lane < 16) ? 0 : 8, kbB = (lane < 16) ? 0 : 8;
  int r0 = wave * 16;

  v8f acc[8];
#pragma unroll
  for (int nt = 0; nt < 8; ++nt) { v8f z = ZERO8; acc[nt] = z; }
#pragma unroll
  for (int kt = 0; kt < 4; ++kt) {
    Frag16 a;
    int c0 = kt * 32 + kbA;
    a.lo = *reinterpret_cast<const v8u*>(&mq[(r0 + row) * 128 + c0]);
    a.hi = *reinterpret_cast<const v8u*>(&mq[(r0 + row) * 128 + c0 + 16]);
    int k0 = kt * 32 + kbB;
    Frag16 bfr[8];
#pragma unroll
    for (int nt = 0; nt < 8; ++nt) {
      int nn = nt * 16 + (lane & 15);
      bfr[nt].lo = *reinterpret_cast<const v8u*>(&mc[nn * 128 + k0]);
      bfr[nt].hi = *reinterpret_cast<const v8u*>(&mc[nn * 128 + k0 + 16]);
    }
#pragma unroll
    for (int nt = 0; nt < 8; ++nt) acc[nt] = wmma_bf16(a, bfr[nt], acc[nt]);
  }
#pragma unroll
  for (int nt = 0; nt < 8; ++nt) {
#pragma unroll
    for (int j = 0; j < 8; ++j) {
      int rr = r0 + j + ((lane < 16) ? 0 : 8);
      int c  = nt * 16 + (lane & 15);
      sim[(long)p * (MAX_SET * MAX_SET) + rr * 128 + c] = acc[nt][j];
    }
  }
}

// ---------------------------------------------------------------------------
// Sinkhorn: one block (128 threads) per pair, 64KB LDS tile.
// Row pass rotates the column index by lane -> conflict free both passes.
// Writes plan in place and masked row sums (temp_mask).
// ---------------------------------------------------------------------------
__global__ void __launch_bounds__(128, 1)
sinkhorn_kernel(float* __restrict__ plan, float* __restrict__ temp_mask) {
  __shared__ float la[MAX_SET * MAX_SET];   // 65536 B
  int p = blockIdx.x, t = threadIdx.x;
  float* g = &plan[(long)p * (MAX_SET * MAX_SET)];
  for (int i = t; i < MAX_SET * MAX_SET; i += 128) la[i] = g[i] * TEMP_INV;
  __syncthreads();
  for (int it = 0; it < SINK_ITERS; ++it) {
    { // row logsumexp (axis=2)
      int q = t; float m = -3.4e38f;
      for (int i = 0; i < 128; ++i) { int c = (t + i) & 127; m = fmaxf(m, la[q * 128 + c]); }
      float s = 0.f;
      for (int i = 0; i < 128; ++i) { int c = (t + i) & 127; s += __expf(la[q * 128 + c] - m); }
      float l = m + __logf(s);
      for (int i = 0; i < 128; ++i) { int c = (t + i) & 127; la[q * 128 + c] -= l; }
    }
    __syncthreads();
    { // col logsumexp (axis=1)
      int c = t; float m = -3.4e38f;
      for (int q = 0; q < 128; ++q) m = fmaxf(m, la[q * 128 + c]);
      float s = 0.f;
      for (int q = 0; q < 128; ++q) s += __expf(la[q * 128 + c] - m);
      float l = m + __logf(s);
      for (int q = 0; q < 128; ++q) la[q * 128 + c] -= l;
    }
    __syncthreads();
  }
  for (int i = t; i < MAX_SET * MAX_SET; i += 128) la[i] = __expf(la[i]);
  __syncthreads();
  { // masked row sums
    int q = t; float s = 0.f;
    for (int i = 0; i < 128; ++i) { int c = (t + i) & 127; s += la[q * 128 + c]; }
    int qsz = 96 + (26 * p) % 33;             // SIZES[2p]
    temp_mask[p * 128 + q] = (q < qsz) ? s : 0.f;
  }
  for (int i = t; i < MAX_SET * MAX_SET; i += 128) g[i] = la[i];
}

__global__ void mask_update_kernel(const int* node_slot, const int* node_pos,
                                   const float* temp_mask, float* mask, int N) {
  long n = (long)blockIdx.x * blockDim.x + threadIdx.x;
  if (n >= N) return;
  int slot = node_slot[n];
  mask[n] = (slot & 1) ? temp_mask[(slot >> 1) * 128 + node_pos[n]] : 1.0f;
}

// ---------------------------------------------------------------------------
// scores[p] = -sum relu(q_emb - plan @ c_emb). One block (8 waves) per pair.
// c_emb staged transposed in LDS so B fragments are contiguous.
// ---------------------------------------------------------------------------
__global__ void __launch_bounds__(256, 1)
score_kernel(const float* __restrict__ plan,
             const float* __restrict__ stacked,
             const u16* __restrict__ stackedb,
             float* __restrict__ out) {
  __shared__ u16 cT[64 * 136];
  __shared__ float red[256];
  int p = blockIdx.x, tid = threadIdx.x;
  int lane = tid & 31, wave = tid >> 5;
  const u16* cb = &stackedb[(long)(2 * p + 1) * MAX_SET * DD];
  for (int i = tid; i < MAX_SET * DD; i += 256) {
    int c = i >> 6, d = i & 63;
    cT[d * 136 + c] = cb[i];
  }
  __syncthreads();
  const float* pl = &plan[(long)p * (MAX_SET * MAX_SET)];
  const float* qe = &stacked[(long)(2 * p) * MAX_SET * DD];
  int row = lane & 15, kbA = (lane < 16) ? 0 : 8, kbB = (lane < 16) ? 0 : 8;
  int r0 = wave * 16;
  float local = 0.f;

  v8f acc[4];
#pragma unroll
  for (int nt = 0; nt < 4; ++nt) { v8f z = ZERO8; acc[nt] = z; }
#pragma unroll
  for (int kt = 0; kt < 4; ++kt) {
    Frag16 a;
    int c0 = kt * 32 + kbA;
    a.lo = cvt8(&pl[(r0 + row) * 128 + c0]);
    a.hi = cvt8(&pl[(r0 + row) * 128 + c0 + 16]);
    int k0 = kt * 32 + kbB;
    Frag16 bfr[4];
#pragma unroll
    for (int nt = 0; nt < 4; ++nt) {
      int nn = nt * 16 + (lane & 15);
      bfr[nt].lo = *reinterpret_cast<const v8u*>(&cT[nn * 136 + k0]);
      bfr[nt].hi = *reinterpret_cast<const v8u*>(&cT[nn * 136 + k0 + 16]);
    }
#pragma unroll
    for (int nt = 0; nt < 4; ++nt) acc[nt] = wmma_bf16(a, bfr[nt], acc[nt]);
  }
#pragma unroll
  for (int nt = 0; nt < 4; ++nt) {
#pragma unroll
    for (int j = 0; j < 8; ++j) {
      int rr = r0 + j + ((lane < 16) ? 0 : 8);
      int c  = nt * 16 + (lane & 15);
      local += fmaxf(qe[rr * 64 + c] - acc[nt][j], 0.f);
    }
  }
  red[tid] = local;
  __syncthreads();
  for (int s = 128; s > 0; s >>= 1) {
    if (tid < s) red[tid] += red[tid + s];
    __syncthreads();
  }
  if (tid == 0) out[p] = -red[0];
}

// ---------------------------------------------------------------------------
// Host-side orchestration
// ---------------------------------------------------------------------------
static inline long cdiv(long a, long b) { return (a + b - 1) / b; }

extern "C" void kernel_launch(void* const* d_in, const int* in_sizes, int n_in,
                              void* d_out, int out_size, void* d_ws, size_t ws_size,
                              hipStream_t stream) {
  (void)n_in; (void)out_size; (void)ws_size;
  const float* nf   = (const float*)d_in[0];
  const float* ef   = (const float*)d_in[1];
  const float* W_ne = (const float*)d_in[2];
  const float* b_ne = (const float*)d_in[3];
  const float* W_ee = (const float*)d_in[4];
  const float* b_ee = (const float*)d_in[5];
  const float* W_m1 = (const float*)d_in[6];
  const float* b_m1 = (const float*)d_in[7];
  const float* W_m2 = (const float*)d_in[8];
  const float* b_m2 = (const float*)d_in[9];
  const float* W_ih = (const float*)d_in[10];
  const float* b_ih = (const float*)d_in[11];
  const float* W_hh = (const float*)d_in[12];
  const float* b_hh = (const float*)d_in[13];
  const float* W_f1 = (const float*)d_in[14];
  const float* b_f1 = (const float*)d_in[15];
  const float* W_f2 = (const float*)d_in[16];
  const float* b_f2 = (const float*)d_in[17];
  const int* from_idx = (const int*)d_in[18];
  const int* to_idx   = (const int*)d_in[19];

  const int N = in_sizes[0] / 32;
  const int E = in_sizes[1] / 16;

  // ---- workspace layout (bump allocator, 256B aligned) ----
  char* wp = (char*)d_ws;
  auto alloc = [&](size_t bytes) -> void* {
    void* r = (void*)wp;
    wp += (bytes + 255) & ~(size_t)255;
    return r;
  };
  int*   offsets   = (int*)  alloc(256 * 4);
  int*   node_slot = (int*)  alloc((size_t)N * 4);
  int*   node_pos  = (int*)  alloc((size_t)N * 4);
  float* mask      = (float*)alloc((size_t)N * 4);
  float* h_enc     = (float*)alloc((size_t)N * 64 * 4);
  u16*   hb        = (u16*)  alloc((size_t)N * 64 * 2);
  u16*   ebf       = (u16*)  alloc((size_t)E * 64 * 2);
  u16*   Wm1s      = (u16*)  alloc(6 * 8 * 512 * 2);
  u16*   Wm2s      = (u16*)  alloc(4 * 8 * 512 * 2);
  float* Wcat      = (float*)alloc(192 * 256 * 4);
  float* b_cat     = (float*)alloc(256 * 4);
  u16*   Wcats     = (u16*)  alloc(6 * 16 * 512 * 2);
  u16*   Wf1s      = (u16*)  alloc(2 * 8 * 512 * 2);
  u16*   Wf2s      = (u16*)  alloc(4 * 8 * 512 * 2);
  float* agg       = (float*)alloc((size_t)N * 128 * 4);
  float* stacked   = (float*)alloc((size_t)NUM_GRAPHS * MAX_SET * DD * 4);
  u16*   stackedb  = (u16*)  alloc((size_t)NUM_GRAPHS * MAX_SET * DD * 2);
  u16*   mqc       = (u16*)  alloc((size_t)NUM_GRAPHS * MAX_SET * MAX_SET * 2);
  float* plan      = (float*)alloc((size_t)128 * MAX_SET * MAX_SET * 4);
  float* temp_mask = (float*)alloc(128 * 128 * 4);

  // ---- one-time prep ----
  meta_kernel<<<1, 256, 0, stream>>>(offsets, node_slot, node_pos, mask, N);
  enc_nodes_kernel<<<(int)cdiv((long)N * 64, 256), 256, 0, stream>>>(nf, W_ne, b_ne, h_enc, hb, (long)N * 64);
  enc_edges_kernel<<<(int)cdiv((long)E * 64, 256), 256, 0, stream>>>(ef, W_ee, b_ee, ebf, (long)E * 64);
  build_wcat_kernel<<<192, 256, 0, stream>>>(W_ih, W_hh, b_ih, b_hh, Wcat, b_cat);
  swizzle_kernel<<<(int)cdiv(6 * 8 * 512, 256), 256, 0, stream>>>(W_m1, Wm1s, 192, 128);
  swizzle_kernel<<<(int)cdiv(4 * 8 * 512, 256), 256, 0, stream>>>(W_m2, Wm2s, 128, 128);
  swizzle_kernel<<<(int)cdiv(6 * 16 * 512, 256), 256, 0, stream>>>(Wcat, Wcats, 192, 256);
  swizzle_kernel<<<(int)cdiv(2 * 8 * 512, 256), 256, 0, stream>>>(W_f1, Wf1s, 64, 128);
  swizzle_kernel<<<(int)cdiv(4 * 8 * 512, 256), 256, 0, stream>>>(W_f2, Wf2s, 128, 128);

  const int etiles = (int)cdiv(E, 16);
  const int eblocks = (int)cdiv(etiles, 8);
  const int ntiles = (int)cdiv(N, 16);
  const int nblocks = (int)cdiv(ntiles, 8);

  // ---- outer iterations ----
  for (int it = 0; it < OUTER_ITERS; ++it) {
    zero_f32_kernel<<<(int)cdiv((long)N * 128, 256), 256, 0, stream>>>(agg, (long)N * 128);
    zero_f32_kernel<<<(int)cdiv((long)NUM_GRAPHS * MAX_SET * DD, 256), 256, 0, stream>>>(
        stacked, (long)NUM_GRAPHS * MAX_SET * DD);
    zero_u16_kernel<<<(int)cdiv((long)NUM_GRAPHS * MAX_SET * DD, 256), 256, 0, stream>>>(
        stackedb, (long)NUM_GRAPHS * MAX_SET * DD);

    edge_mlp_kernel<<<dim3(eblocks, 2), 256, 0, stream>>>(
        hb, ebf, Wm1s, Wm2s, b_m1, b_m2, mask, from_idx, to_idx, agg, E);

    gru_kernel<<<nblocks, 256, 0, stream>>>(
        agg, hb, h_enc, Wcats, b_cat, node_slot, node_pos, stacked, stackedb, N);

    tqtc_kernel<<<NUM_GRAPHS, 256, 0, stream>>>(stackedb, Wf1s, Wf2s, b_f1, b_f2, mqc);

    sim_kernel<<<128, 256, 0, stream>>>(mqc, plan);

    sinkhorn_kernel<<<128, 128, 0, stream>>>(plan, temp_mask);

    mask_update_kernel<<<(int)cdiv(N, 256), 256, 0, stream>>>(
        node_slot, node_pos, temp_mask, mask, N);
  }

  score_kernel<<<128, 256, 0, stream>>>(plan, stacked, stackedb, (float*)d_out);
}
